// DeepseekV2MoE_44616120271590
// MI455X (gfx1250) — compile-verified, matrix operand order checked by reading, not run on hardware
//
#include <hip/hip_runtime.h>

typedef __attribute__((ext_vector_type(16))) __bf16 v16bf;
typedef __attribute__((ext_vector_type(8)))  float  v8f;
typedef __attribute__((ext_vector_type(4)))  unsigned int v4u;
typedef __attribute__((ext_vector_type(8)))  unsigned int v8u;

// ---- Problem constants (DeepSeek-V2 style MoE) ----
constexpr int Tn  = 1024;   // tokens = B*S
constexpr int Hn  = 2048;   // hidden
constexpr int In  = 1408;   // moe intermediate
constexpr int En  = 32;     // routed experts
constexpr int Kn  = 6;      // experts per token
constexpr int Gn  = 8;      // routing groups
constexpr int TGn = 3;      // top-k groups
constexpr int NSH = 2;      // shared experts
constexpr int Cn  = 384;    // capacity per expert
constexpr float SCALEW = 16.0f;
constexpr int BK  = 32;     // WMMA K step for bf16 (16x16x32)

// ---------------------------------------------------------------------------
// 0) init: slot_of = -1 (dropped / unassigned sentinel)
// ---------------------------------------------------------------------------
__global__ void init_slots_kernel(int* __restrict__ slot_of) {
  int i = blockIdx.x * blockDim.x + threadIdx.x;
  if (i < Tn * Kn) slot_of[i] = -1;
}

// ---------------------------------------------------------------------------
// 1) MoE gate: logits, group-limited greedy top-3 groups, softmax (masked->0.0
//    entries INCLUDED in softmax, matching reference), top-6, *16 scale.
// ---------------------------------------------------------------------------
__global__ __launch_bounds__(256) void gate_topk_kernel(
    const float* __restrict__ x, const float* __restrict__ gw,
    int* __restrict__ topk_idx, float* __restrict__ topk_w) {
  __shared__ float logits[8][En];
  const int lane = threadIdx.x & 31;
  const int wv   = threadIdx.x >> 5;
  const int t    = blockIdx.x * 8 + wv;

  const float* xr = x  + (size_t)t * Hn;
  const float* wr = gw + (size_t)lane * Hn;
  float acc = 0.f;
  #pragma unroll 4
  for (int h = 0; h < Hn; ++h) acc += xr[h] * wr[h];
  logits[wv][lane] = acc;
  __syncthreads();

  if (lane == 0) {
    float sc[En];
    #pragma unroll
    for (int e = 0; e < En; ++e) sc[e] = logits[wv][e];

    float gsc[Gn];
    for (int g = 0; g < Gn; ++g) {
      float m = sc[g * 4];
      for (int j = 1; j < 4; ++j) m = fmaxf(m, sc[g * 4 + j]);
      gsc[g] = m;
    }
    bool keep[Gn] = {};
    for (int r = 0; r < TGn; ++r) {
      int bi = 0; float bv = -3.4e38f;
      for (int g = 0; g < Gn; ++g)
        if (!keep[g] && gsc[g] > bv) { bv = gsc[g]; bi = g; }
      keep[bi] = true;
    }
    float ms[En]; float mx = -3.4e38f;
    for (int e = 0; e < En; ++e) {
      ms[e] = keep[e >> 2] ? sc[e] : 0.0f;   // masked_fill(~mask, 0.0)
      mx = fmaxf(mx, ms[e]);
    }
    float sum = 0.f;
    for (int e = 0; e < En; ++e) { ms[e] = __expf(ms[e] - mx); sum += ms[e]; }
    float inv = 1.f / sum;
    for (int e = 0; e < En; ++e) ms[e] *= inv;
    for (int k = 0; k < Kn; ++k) {
      int bi = 0; float bv = -1.f;
      for (int e = 0; e < En; ++e) if (ms[e] > bv) { bv = ms[e]; bi = e; }
      ms[bi] = -1.f;
      topk_idx[t * Kn + k] = bi;
      topk_w[t * Kn + k]   = bv * SCALEW;    // renormalize=False branch
    }
  }
}

// ---------------------------------------------------------------------------
// 2) Deterministic capacity dispatch: one block per expert scans the T*K
//    assignments in order (matches reference cumsum) with wave32 ballots.
// ---------------------------------------------------------------------------
__global__ __launch_bounds__(256) void dispatch_kernel(
    const int* __restrict__ topk_idx, int* __restrict__ row_token,
    int* __restrict__ slot_of) {
  const int e    = blockIdx.x;
  const int lane = threadIdx.x & 31;
  const int wv   = threadIdx.x >> 5;
  __shared__ int wave_cnt[8];
  __shared__ int wave_off[8];
  __shared__ int running;
  if (threadIdx.x == 0) running = 0;
  __syncthreads();

  const int TK = Tn * Kn;                 // 6144 = 24 * 256 exactly
  for (int base = 0; base < TK; base += 256) {
    const int  i     = base + threadIdx.x;
    const bool match = (topk_idx[i] == e);
    unsigned mask = __builtin_amdgcn_ballot_w32(match);
    if (lane == 0) wave_cnt[wv] = __builtin_popcount(mask);
    __syncthreads();
    if (threadIdx.x == 0) {
      int s = running;
      #pragma unroll
      for (int w = 0; w < 8; ++w) { wave_off[w] = s; s += wave_cnt[w]; }
      running = s;
    }
    __syncthreads();
    if (match) {
      int pos = wave_off[wv] + __builtin_popcount(mask & ((1u << lane) - 1u));
      if (pos < Cn) {
        row_token[e * Cn + pos] = i / Kn;
        slot_of[i] = e * Cn + pos;
      }
    }
    __syncthreads();
  }
  int cnt = running < Cn ? running : Cn;
  for (int r = cnt + threadIdx.x; r < Cn; r += 256) row_token[e * Cn + r] = -1;
}

// ---------------------------------------------------------------------------
// 3) Fused gate_up GEMM + SiLU*mul (bf16 WMMA, fp32 accum).
//    Block tile: M=128, N=64 (gate half and up half simultaneously).
//    KDIM/IHALF are compile-time so weight loads fold to immediate offsets,
//    and a full wave reads 32 consecutive floats per k-row (128B coalesced).
// ---------------------------------------------------------------------------
template <bool GATHER, int KDIM, int IHALF>
__global__ __launch_bounds__(256) void gateup_silu_kernel(
    const float* __restrict__ A,       // [Tn, KDIM] token activations (fp32)
    const float* __restrict__ W,       // [gridDim.z, KDIM, 2*IHALF] fp32
    const int* __restrict__ row_tok,   // [gridDim.z * Cn] or nullptr
    __bf16* __restrict__ out) {        // [rows, IHALF] bf16 activations
  constexpr int LDW = 2 * IHALF;
  const int m0      = blockIdx.y * 128;
  const int n0      = blockIdx.x * 64;
  const int rowbase = GATHER ? (int)blockIdx.z * Cn : 0;
  const float* w    = W + (size_t)blockIdx.z * KDIM * LDW;

  __shared__ __align__(32) __bf16 lA[128 * BK];   // [row][k]
  __shared__ __align__(32) __bf16 lG[64 * BK];    // [col][k] (transposed B)
  __shared__ __align__(32) __bf16 lU[64 * BK];

  const int tid  = threadIdx.x;
  const int lane = tid & 31;
  const int wv   = tid >> 5;
  const int wm   = (wv & 3) * 32;   // wave M offset in tile
  const int wn   = (wv >> 2) * 32;  // wave N offset in tile

  v8f accg[2][2] = {};
  v8f accu[2][2] = {};

  // A staging: each thread converts 16 fp32 -> bf16 for half a row
  const int arow = tid >> 1;
  const int acol = (tid & 1) * 16;
  const float* asrc;
  if (GATHER) {
    int tok = row_tok[rowbase + m0 + arow];
    asrc = (tok >= 0) ? (A + (size_t)tok * KDIM) : nullptr;  // empty row -> 0
  } else {
    asrc = A + (size_t)(m0 + arow) * KDIM;
  }
  // B staging: wave-coalesced: consecutive tids -> consecutive columns
  const int bcol = tid & 63;        // 0..63
  const int bkb  = tid >> 6;        // 0..3 (k base; k = bkb + 4*j)
  const float* bbase = w + (size_t)bkb * LDW + (n0 + bcol);

  for (int k0 = 0; k0 < KDIM; k0 += BK) {
    __bf16* da = &lA[arow * BK + acol];
    if (asrc) {
      const float* s = asrc + k0 + acol;
      #pragma unroll
      for (int j = 0; j < 16; ++j) da[j] = (__bf16)s[j];
    } else {
      #pragma unroll
      for (int j = 0; j < 16; ++j) da[j] = (__bf16)0.f;
    }
    const float* gs = bbase + (size_t)k0 * LDW;
    #pragma unroll
    for (int j = 0; j < 8; ++j) {          // compile-time immediate offsets
      lG[bcol * BK + bkb + 4 * j] = (__bf16)gs[(size_t)(4 * j) * LDW];
      lU[bcol * BK + bkb + 4 * j] = (__bf16)gs[(size_t)(4 * j) * LDW + IHALF];
    }
    __syncthreads();
    if (k0 + BK < KDIM)
      __builtin_prefetch(bbase + (size_t)(k0 + BK) * LDW, 0, 1); // global_prefetch_b8

    v16bf af[2], gf[2], uf[2];
    #pragma unroll
    for (int i = 0; i < 2; ++i) {
      af[i] = *(const v16bf*)&lA[(wm + i * 16 + (lane & 15)) * BK + (lane >> 4) * 16];
      gf[i] = *(const v16bf*)&lG[(wn + i * 16 + (lane & 15)) * BK + (lane >> 4) * 16];
      uf[i] = *(const v16bf*)&lU[(wn + i * 16 + (lane & 15)) * BK + (lane >> 4) * 16];
    }
    #pragma unroll
    for (int i = 0; i < 2; ++i)
      #pragma unroll
      for (int j = 0; j < 2; ++j) {
        accg[i][j] = __builtin_amdgcn_wmma_f32_16x16x32_bf16(
            false, af[i], false, gf[j], (short)0, accg[i][j], false, false);
        accu[i][j] = __builtin_amdgcn_wmma_f32_16x16x32_bf16(
            false, af[i], false, uf[j], (short)0, accu[i][j], false, false);
      }
    __syncthreads();
  }

  // Epilogue: act = silu(g) * u, stored bf16.
  // C/D layout: VGPR r, lanes 0-15 -> M=r, lanes 16-31 -> M=8+r; N = lane&15.
  #pragma unroll
  for (int i = 0; i < 2; ++i)
    #pragma unroll
    for (int j = 0; j < 2; ++j)
      #pragma unroll
      for (int r = 0; r < 8; ++r) {
        int m = m0 + wm + i * 16 + r + ((lane >> 4) * 8);
        int n = n0 + wn + j * 16 + (lane & 15);
        float g = accg[i][j][r];
        float u = accu[i][j][r];
        float a = (g / (1.f + __expf(-g))) * u;
        out[(size_t)(rowbase + m) * IHALF + n] = (__bf16)a;
      }
}

// ---------------------------------------------------------------------------
// 4) Down-projection GEMM (bf16 WMMA, fp32 accum). Block tile 128x128.
//    The A tile (bf16 [128 x 32] row-major slab) is moved HBM->LDS by the
//    Tensor Data Mover: a 2-group D# (ISA 08_async_tensor 8.3/8.4) issued
//    once per k-step from wave 0 via inline asm (TDM ignores EXEC), completed
//    with s_wait_tensorcnt + workgroup barrier.
// ---------------------------------------------------------------------------
template <int KDIM, int NLD, int ROWS_PER_Z, int TOTROWS, bool OUT_BF16>
__global__ __launch_bounds__(256) void down_kernel(
    const __bf16* __restrict__ A,     // [TOTROWS, KDIM] bf16 activations
    const float* __restrict__ W,      // [gridDim.z, KDIM, NLD] fp32
    __bf16* __restrict__ out_bf,
    float* __restrict__ out_f32) {
  const int m0      = blockIdx.y * 128;
  const int n0      = blockIdx.x * 128;
  const int rowbase = blockIdx.z * ROWS_PER_Z;
  const float* w    = W + (size_t)blockIdx.z * KDIM * NLD;

  __shared__ __align__(32) __bf16 lA[128 * BK];   // [row][k] -- filled by TDM
  __shared__ __align__(32) __bf16 lB[128 * BK];   // [col][k]

  const int tid  = threadIdx.x;
  const int lane = tid & 31;
  const int wv   = tid >> 5;
  const int wm   = (wv & 3) * 32;
  const int wn   = (wv >> 2) * 64;

  v8f acc[2][4] = {};

  const int bcol = tid & 127;       // 0..127: consecutive tids -> cols (coalesced)
  const int bkb  = tid >> 7;        // 0..1  (k = bkb + 2*j)
  const float* bbase = w + (size_t)bkb * NLD + (n0 + bcol);

  const unsigned lds_base = (unsigned)(size_t)(void*)lA;           // LDS byte offset
  const unsigned long long gtile =
      (unsigned long long)(size_t)(A + (size_t)(rowbase + m0) * KDIM);

  for (int k0 = 0; k0 < KDIM; k0 += BK) {
    // --- TDM: async-load the 128x32 bf16 A slab into LDS (one issue/block) ---
    if (tid == 0) {
      const unsigned long long ga = gtile + (unsigned long long)k0 * 2ull;
      v4u g0;
      g0[0] = 1u;                                    // count=1, user descriptor
      g0[1] = lds_base;                              // lds_addr (bytes)
      g0[2] = (unsigned)ga;                          // global_addr[31:0]
      g0[3] = ((unsigned)(ga >> 32) & 0x01FFFFFFu)   // global_addr[56:32]
              | (2u << 30);                          // type = 2 ("image")
      v8u g1;
      g1[0] = 1u << 16;                              // data_size=1 (2 bytes/elem)
      g1[1] = ((unsigned)KDIM & 0xFFFFu) << 16;      // tensor_dim0[15:0]
      g1[2] = ((unsigned)KDIM >> 16)                 // tensor_dim0[31:16]
              | (((unsigned)TOTROWS & 0xFFFFu) << 16); // tensor_dim1[15:0]
      g1[3] = ((unsigned)TOTROWS >> 16)              // tensor_dim1[31:16]
              | ((unsigned)BK << 16);                // tile_dim0 = 32 elems
      g1[4] = 128u;                                  // tile_dim1=128, tile_dim2=0
      g1[5] = (unsigned)KDIM;                        // tensor_dim0_stride[31:0]
      g1[6] = 0u;                                    // stride hi | dim1_stride lo
      g1[7] = 0u;
      asm volatile("tensor_load_to_lds %0, %1" : : "s"(g0), "s"(g1) : "memory");
    }
    // --- B stage: fp32 weights -> bf16, [col][k] in LDS (immediate offsets) ---
    const float* bs = bbase + (size_t)k0 * NLD;
    #pragma unroll
    for (int j = 0; j < 16; ++j)
      lB[bcol * BK + bkb + 2 * j] = (__bf16)bs[(size_t)(2 * j) * NLD];

    if (wv == 0) __builtin_amdgcn_s_wait_tensorcnt(0);  // A slab landed in LDS
    __syncthreads();
    if (k0 + BK < KDIM)
      __builtin_prefetch(bbase + (size_t)(k0 + BK) * NLD, 0, 1);

    v16bf af[2], bfr[4];
    #pragma unroll
    for (int i = 0; i < 2; ++i)
      af[i] = *(const v16bf*)&lA[(wm + i * 16 + (lane & 15)) * BK + (lane >> 4) * 16];
    #pragma unroll
    for (int j = 0; j < 4; ++j)
      bfr[j] = *(const v16bf*)&lB[(wn + j * 16 + (lane & 15)) * BK + (lane >> 4) * 16];
    #pragma unroll
    for (int i = 0; i < 2; ++i)
      #pragma unroll
      for (int j = 0; j < 4; ++j)
        acc[i][j] = __builtin_amdgcn_wmma_f32_16x16x32_bf16(
            false, af[i], false, bfr[j], (short)0, acc[i][j], false, false);
    __syncthreads();   // protect lA/lB before next TDM/stage
  }

  #pragma unroll
  for (int i = 0; i < 2; ++i)
    #pragma unroll
    for (int j = 0; j < 4; ++j)
      #pragma unroll
      for (int r = 0; r < 8; ++r) {
        int m = m0 + wm + i * 16 + r + ((lane >> 4) * 8);
        int n = n0 + wn + j * 16 + (lane & 15);
        size_t idx = (size_t)(rowbase + m) * NLD + n;
        float v = acc[i][j][r];
        if constexpr (OUT_BF16) out_bf[idx] = (__bf16)v;
        else                    out_f32[idx] += v;
      }
}

// ---------------------------------------------------------------------------
// 5) Combine: out[t,:] = sum_k w_tk * yb[slot_tk, :]  (dropped slots skipped)
// ---------------------------------------------------------------------------
__global__ __launch_bounds__(256) void combine_kernel(
    const __bf16* __restrict__ yb, const int* __restrict__ slot_of,
    const float* __restrict__ topk_w, float* __restrict__ out) {
  const int t = blockIdx.x;
  __shared__ int   sl[Kn];
  __shared__ float wt[Kn];
  if (threadIdx.x < Kn) {
    sl[threadIdx.x] = slot_of[t * Kn + threadIdx.x];
    wt[threadIdx.x] = topk_w[t * Kn + threadIdx.x];
  }
  __syncthreads();
  for (int h = threadIdx.x; h < Hn; h += 256) {
    float s = 0.f;
    #pragma unroll
    for (int k = 0; k < Kn; ++k) {
      int slot = sl[k];
      if (slot >= 0) s += wt[k] * (float)yb[(size_t)slot * Hn + h];
    }
    out[(size_t)t * Hn + h] = s;
  }
}

// ---------------------------------------------------------------------------
// Launch
// ---------------------------------------------------------------------------
extern "C" void kernel_launch(void* const* d_in, const int* in_sizes, int n_in,
                              void* d_out, int out_size, void* d_ws, size_t ws_size,
                              hipStream_t stream) {
  const float* x       = (const float*)d_in[0];  // [1,1024,2048]
  const float* gate_w  = (const float*)d_in[1];  // [32,2048]
  const float* w_gu    = (const float*)d_in[2];  // [32,2048,2816]
  const float* w_down  = (const float*)d_in[3];  // [32,1408,2048]
  const float* ws_gu   = (const float*)d_in[4];  // [2048,5632]
  const float* ws_down = (const float*)d_in[5];  // [2816,2048]
  float* out = (float*)d_out;                    // [1,1024,2048] fp32

  char* ws = (char*)d_ws;
  size_t off = 0;
  auto take = [&](size_t bytes) { char* p = ws + off; off = (off + bytes + 255) & ~(size_t)255; return p; };
  int*    topk_idx  = (int*)  take((size_t)Tn * Kn * 4);
  float*  topk_w    = (float*)take((size_t)Tn * Kn * 4);
  int*    row_token = (int*)  take((size_t)En * Cn * 4);
  int*    slot_of   = (int*)  take((size_t)Tn * Kn * 4);
  __bf16* act       = (__bf16*)take((size_t)En * Cn * In * 2);        // 34.6 MB
  __bf16* yb        = (__bf16*)take((size_t)En * Cn * Hn * 2);        // 50.3 MB
  __bf16* sact      = (__bf16*)take((size_t)Tn * (In * NSH) * 2);     //  5.8 MB

  // 0) slot sentinel init
  init_slots_kernel<<<(Tn * Kn + 255) / 256, 256, 0, stream>>>(slot_of);
  // 1) routing
  gate_topk_kernel<<<Tn / 8, 256, 0, stream>>>(x, gate_w, topk_idx, topk_w);
  // 2) capacity dispatch
  dispatch_kernel<<<En, 256, 0, stream>>>(topk_idx, row_token, slot_of);
  // 3) routed experts: gate_up + SiLU*mul  (grid: N-tiles x M-tiles x experts)
  gateup_silu_kernel<true, Hn, In><<<dim3(In / 64, Cn / 128, En), 256, 0, stream>>>(
      x, w_gu, row_token, act);
  // 4) routed experts: down projection -> yb (bf16), A tile via TDM
  down_kernel<In, Hn, Cn, En * Cn, true><<<dim3(Hn / 128, Cn / 128, En), 256, 0, stream>>>(
      act, w_down, yb, nullptr);
  // 5) shared experts: gate_up + SiLU*mul over all tokens
  gateup_silu_kernel<false, Hn, In * NSH><<<dim3((In * NSH) / 64, Tn / 128, 1), 256, 0, stream>>>(
      x, ws_gu, nullptr, sact);
  // 6) weighted combine of routed outputs (writes every out element)
  combine_kernel<<<Tn, 256, 0, stream>>>(yb, slot_of, topk_w, out);
  // 7) shared experts: down projection, accumulated onto out (after combine)
  down_kernel<In * NSH, Hn, 0, Tn, false><<<dim3(Hn / 128, Tn / 128, 1), 256, 0, stream>>>(
      sact, ws_down, nullptr, out);
}
